// KNNClassifier_4741643895153
// MI455X (gfx1250) — matrix operand order repaired and checked by reading, not compiled, not run on hardware
//
#include <hip/hip_runtime.h>
#include <stdint.h>

typedef __attribute__((ext_vector_type(16))) __bf16 v16bf;
typedef __attribute__((ext_vector_type(8)))  float  v8f;
typedef unsigned int v4u __attribute__((ext_vector_type(4)));
typedef int          v8i __attribute__((ext_vector_type(8)));
typedef int          v4i __attribute__((ext_vector_type(4)));

#define DIMS        512
#define NTRAIN      65536
#define NQ          2048
#define KNN         5
#define NSPLIT      32
#define LSTR        520         // padded LDS row stride (ushorts) -> conflict-free ds_load_b128

// fallback-path geometry (on-the-fly conversion)
#define FB_QB       32
#define FB_NBLK     16
#define FB_TILES    128         // 2048 rows per stream / 16
// fast-path geometry (preconverted bf16 + TDM)
#define FP_QB       64
#define FP_NBLK     32
#define FP_TILES    128         // 2048 rows per block / 16

#if defined(__has_builtin)
#if __has_builtin(__builtin_amdgcn_tensor_load_to_lds) && __has_builtin(__builtin_amdgcn_s_wait_tensorcnt)
#define USE_TDM 1
#endif
#endif
#ifndef USE_TDM
#define USE_TDM 0
#endif

// ---- bf16 helpers (round-to-nearest-even) ----
__device__ __forceinline__ unsigned short f2bf(float f) {
    unsigned u = __float_as_uint(f);
    unsigned r = u + 0x7FFFu + ((u >> 16) & 1u);
    return (unsigned short)(r >> 16);
}

__device__ __forceinline__ v16bf ld_frag(const unsigned short* base, int o0, int o1) {
    union { uint4 q[2]; v16bf v; } f;
    f.q[0] = *(const uint4*)(base + o0);
    f.q[1] = *(const uint4*)(base + o1);
    return f.v;
}

// sorted-5 lexicographic insert
__device__ __forceinline__ void ins5(float d, int ix, float td[5], int ti[5]) {
    if (d < td[4] || (d == td[4] && ix < ti[4])) {
        td[4] = d; ti[4] = ix;
#pragma unroll
        for (int u = 4; u >= 1; --u) {
            if (td[u] < td[u - 1] || (td[u] == td[u - 1] && ti[u] < ti[u - 1])) {
                float t0 = td[u]; td[u] = td[u - 1]; td[u - 1] = t0;
                int   i0 = ti[u]; ti[u] = ti[u - 1]; ti[u - 1] = i0;
            }
        }
    }
}

#if USE_TDM
// TDM: move one 16x512 ushort tile (row stride 512 in memory) into LDS with
// 16B padding every 1024B (=> LDS row stride 520 ushorts).
__device__ __forceinline__ void tdm_load_tile(const unsigned short* gsrc, unsigned short* ldst) {
    unsigned long long ga = (unsigned long long)(uintptr_t)gsrc;
    unsigned lds = (unsigned)(uintptr_t)ldst;
    v4u g0;
    g0[0] = 1u;                                                   // count=1, user mode
    g0[1] = lds;                                                  // lds_addr
    g0[2] = (unsigned)ga;                                         // global_addr[31:0]
    g0[3] = (unsigned)((ga >> 32) & 0x01FFFFFFull) | 0x80000000u; // addr[56:32] | type=2
    v8i g1;
    g1[0] = (int)0x07D10000u;      // data_size=2B | pad_enable | pad_interval=256DW | pad_amount=4DW
    g1[1] = (int)(512u << 16);     // tensor_dim0 = 512
    g1[2] = (int)(16u << 16);      // tensor_dim1 = 16
    g1[3] = (int)(512u << 16);     // tile_dim0 = 512
    g1[4] = 16;                    // tile_dim1 = 16
    g1[5] = 512;                   // tensor_dim0_stride = 512
    g1[6] = 0;
    g1[7] = 0;
    v4i z4 = {0, 0, 0, 0};
#if __clang_major__ >= 23
    v8i z8 = {0, 0, 0, 0, 0, 0, 0, 0};
    __builtin_amdgcn_tensor_load_to_lds(g0, g1, z4, z4, z8, 0);
#else
    __builtin_amdgcn_tensor_load_to_lds(g0, g1, z4, z4, 0);
#endif
}
#endif

// ---- kernel: train row norms, one wave per row ----
__global__ __launch_bounds__(256) void train_norms(const float* __restrict__ xt,
                                                   float* __restrict__ tn2) {
    int w = threadIdx.x >> 5, lane = threadIdx.x & 31;
    int row = blockIdx.x * 8 + w;
    const float* p = xt + (size_t)row * DIMS;
    float s = 0.f;
#pragma unroll
    for (int j = 0; j < 4; ++j) {
        float4 v = *(const float4*)&p[lane * 4 + j * 128];
        s += v.x * v.x + v.y * v.y + v.z * v.z + v.w * v.w;
    }
#pragma unroll
    for (int o = 16; o > 0; o >>= 1) s += __shfl_xor(s, o, 32);
    if (lane == 0) tn2[row] = s;
}

// ---- kernel: one-time f32 -> bf16 hi/lo conversion of x_train ----
__global__ __launch_bounds__(256) void conv_train(const float* __restrict__ xt,
                                                  unsigned short* __restrict__ hi,
                                                  unsigned short* __restrict__ lo) {
    size_t e = ((size_t)blockIdx.x * 256 + threadIdx.x) * 4;
    float4 v = *(const float4*)&xt[e];
    unsigned short h0 = f2bf(v.x), h1 = f2bf(v.y), h2 = f2bf(v.z), h3 = f2bf(v.w);
    unsigned short l0 = f2bf(v.x - __uint_as_float((unsigned)h0 << 16));
    unsigned short l1 = f2bf(v.y - __uint_as_float((unsigned)h1 << 16));
    unsigned short l2 = f2bf(v.z - __uint_as_float((unsigned)h2 << 16));
    unsigned short l3 = f2bf(v.w - __uint_as_float((unsigned)h3 << 16));
    uint2 hp; hp.x = (unsigned)h0 | ((unsigned)h1 << 16); hp.y = (unsigned)h2 | ((unsigned)h3 << 16);
    uint2 lp; lp.x = (unsigned)l0 | ((unsigned)l1 << 16); lp.y = (unsigned)l2 | ((unsigned)l3 << 16);
    *(uint2*)&hi[e] = hp;
    *(uint2*)&lo[e] = lp;
}

// =====================================================================
// FAST PATH: preconverted bf16 train data, TDM double-buffered tiles.
// 4 waves, each owns a 16-query tile (64 queries/block); one train tile
// shared by all 4 waves; 128 tiles of 16 rows per block (n-range 2048).
// =====================================================================
__global__ __launch_bounds__(128) void knn_gemm_topk_fast(
    const float* __restrict__ x,
    const unsigned short* __restrict__ thig, const unsigned short* __restrict__ tlog,
    const float* __restrict__ tn2,
    float* __restrict__ candD, int* __restrict__ candI)
{
    __shared__ __align__(16) char smem[199680];
    unsigned short* qhi = (unsigned short*)smem;                   // 64*520*2 = 66560 B
    unsigned short* qlo = (unsigned short*)(smem + 66560);         // 66560 B
    unsigned short* thi = (unsigned short*)(smem + 133120);        // 2 bufs * 16*520*2 = 33280 B
    unsigned short* tlo = (unsigned short*)(smem + 166400);        // 33280 B

    const int tid  = threadIdx.x;
    const int lane = tid & 31;
    const int w    = tid >> 5;
    const int half = lane >> 4;
    const int bq   = blockIdx.x & 31;
    const int nb   = blockIdx.x >> 5;
    const int nstart = nb * 2048;

    // ---- preload + convert this block's 64 queries (once per block) ----
#pragma unroll 4
    for (int i = 0; i < 64; ++i) {
        int e = tid * 4 + i * 512;
        int row = e >> 9, col = e & 511;
        float4 v = *(const float4*)&x[(size_t)(bq * FP_QB + row) * DIMS + col];
        unsigned short h0 = f2bf(v.x), h1 = f2bf(v.y), h2 = f2bf(v.z), h3 = f2bf(v.w);
        unsigned short l0 = f2bf(v.x - __uint_as_float((unsigned)h0 << 16));
        unsigned short l1 = f2bf(v.y - __uint_as_float((unsigned)h1 << 16));
        unsigned short l2 = f2bf(v.z - __uint_as_float((unsigned)h2 << 16));
        unsigned short l3 = f2bf(v.w - __uint_as_float((unsigned)h3 << 16));
        uint2 hp; hp.x = (unsigned)h0 | ((unsigned)h1 << 16); hp.y = (unsigned)h2 | ((unsigned)h3 << 16);
        uint2 lp; lp.x = (unsigned)l0 | ((unsigned)l1 << 16); lp.y = (unsigned)l2 | ((unsigned)l3 << 16);
        *(uint2*)&qhi[row * LSTR + col] = hp;
        *(uint2*)&qlo[row * LSTR + col] = lp;
    }

    const unsigned short* aHi = qhi + (w * 16 + (lane & 15)) * LSTR;
    const unsigned short* aLo = qlo + (w * 16 + (lane & 15)) * LSTR;

    float bd[8][5]; int bi[8][5];
#pragma unroll
    for (int r = 0; r < 8; ++r)
#pragma unroll
        for (int j = 0; j < 5; ++j) { bd[r][j] = __FLT_MAX__; bi[r][j] = 0x7FFFFFFF; }

#if USE_TDM
    // prologue: tile 0 into buffer 0 (wave0: hi, wave1: lo)
    if (w == 0) tdm_load_tile(thig + (size_t)nstart * DIMS, thi);
    if (w == 1) tdm_load_tile(tlog + (size_t)nstart * DIMS, tlo);
#endif

    for (int it = 0; it < FP_TILES; ++it) {
        const int nbase = nstart + it * 16;
        const int buf = it & 1;
#if USE_TDM
        {   // issue next tile into other buffer (last iter: harmless refetch)
            const int itn = (it + 1 < FP_TILES) ? (it + 1) : it;
            if (w == 0) tdm_load_tile(thig + (size_t)(nstart + itn * 16) * DIMS, thi + (buf ^ 1) * 8320);
            if (w == 1) tdm_load_tile(tlog + (size_t)(nstart + itn * 16) * DIMS, tlo + (buf ^ 1) * 8320);
            __builtin_amdgcn_s_wait_tensorcnt(1);  // current tile (oldest) complete
        }
#else
        {   // cooperative bf16 tile copy (no conversion needed)
#pragma unroll
            for (int i = 0; i < 16; ++i) {
                int idx = tid + i * 128;                 // 0..2047 uint4 chunks
                int c = idx & 1023;
                int row = c >> 6, col8 = (c & 63) * 8;
                const unsigned short* src = (idx < 1024 ? thig : tlog) + (size_t)(nbase + row) * DIMS + col8;
                unsigned short* dst = (idx < 1024 ? thi : tlo) + buf * 8320 + row * LSTR + col8;
                *(uint4*)dst = *(const uint4*)src;
            }
        }
#endif
        __syncthreads();

        const unsigned short* bHi = thi + buf * 8320 + (lane & 15) * LSTR;
        const unsigned short* bLo = tlo + buf * 8320 + (lane & 15) * LSTR;

        v8f c = {0.f, 0.f, 0.f, 0.f, 0.f, 0.f, 0.f, 0.f};
#pragma unroll
        for (int k = 0; k < 16; ++k) {
            v16bf ah = ld_frag(aHi, k * 32 + half * 8,  k * 32 + 16 + half * 8);
            v16bf al = ld_frag(aLo, k * 32 + half * 8,  k * 32 + 16 + half * 8);
            v16bf bh = ld_frag(bHi, k * 32 + half * 8,  k * 32 + 16 + half * 8);
            v16bf bl = ld_frag(bLo, k * 32 + half * 8,  k * 32 + 16 + half * 8);
            c = __builtin_amdgcn_wmma_f32_16x16x32_bf16(false, ah, false, bh, (short)0, c, false, false);
            c = __builtin_amdgcn_wmma_f32_16x16x32_bf16(false, ah, false, bl, (short)0, c, false, false);
            c = __builtin_amdgcn_wmma_f32_16x16x32_bf16(false, al, false, bh, (short)0, c, false, false);
        }

        const int nIdx = nbase + (lane & 15);
        const float t2 = tn2[nIdx];
#pragma unroll
        for (int r = 0; r < 8; ++r) {
            float s = fmaf(-2.0f, c[r], t2);
            if (s < bd[r][4]) {
                bd[r][4] = s; bi[r][4] = nIdx;
#pragma unroll
                for (int j = 4; j >= 1; --j) {
                    if (bd[r][j] < bd[r][j - 1]) {
                        float td = bd[r][j]; bd[r][j] = bd[r][j - 1]; bd[r][j - 1] = td;
                        int   ti = bi[r][j]; bi[r][j] = bi[r][j - 1]; bi[r][j - 1] = ti;
                    }
                }
            }
        }
        __syncthreads();
    }

    // ---- merge per-lane top5 within each wave (reuse q LDS region) ----
    float* sd = (float*)smem;
    int*   si = (int*)(smem + 20480);
#pragma unroll
    for (int r = 0; r < 8; ++r)
#pragma unroll
        for (int j = 0; j < 5; ++j) {
            int o = ((w * 32 + lane) * 8 + r) * 5 + j;
            sd[o] = bd[r][j]; si[o] = bi[r][j];
        }
    __syncthreads();

    if (lane < 16) {
        const int m = lane, hh = m >> 3, r = m & 7;
        float td[5]; int ti[5];
#pragma unroll
        for (int j = 0; j < 5; ++j) { td[j] = __FLT_MAX__; ti[j] = 0x7FFFFFFF; }
        for (int sl = 0; sl < 16; ++sl) {
            int L = hh * 16 + sl;
#pragma unroll
            for (int j = 0; j < 5; ++j) {
                int o = ((w * 32 + L) * 8 + r) * 5 + j;
                ins5(sd[o], si[o], td, ti);
            }
        }
        const int qg = bq * FP_QB + w * 16 + m;
#pragma unroll
        for (int j = 0; j < 5; ++j) {
            candD[(qg * NSPLIT + nb) * 5 + j] = td[j];
            candI[(qg * NSPLIT + nb) * 5 + j] = ti[j];
        }
    }
}

// =====================================================================
// FALLBACK PATH: on-the-fly f32->bf16 conversion (round-1 kernel).
// =====================================================================
__global__ __launch_bounds__(128) void knn_gemm_topk_fb(
    const float* __restrict__ x, const float* __restrict__ xt,
    const float* __restrict__ tn2,
    float* __restrict__ candD, int* __restrict__ candI)
{
    __shared__ __align__(16) char smem[133120];
    unsigned short* qhi = (unsigned short*)smem;
    unsigned short* qlo = (unsigned short*)(smem + 33280);
    unsigned short* thi = (unsigned short*)(smem + 66560);
    unsigned short* tlo = (unsigned short*)(smem + 99840);

    const int tid  = threadIdx.x;
    const int lane = tid & 31;
    const int w    = tid >> 5;
    const int half = lane >> 4;
    const int qt   = w & 1;
    const int strm = w >> 1;
    const int bq   = blockIdx.x & 63;
    const int nb   = blockIdx.x >> 6;

#pragma unroll 4
    for (int i = 0; i < 32; ++i) {
        int e = tid * 4 + i * 512;
        int row = e >> 9, col = e & 511;
        float4 v = *(const float4*)&x[(size_t)(bq * FB_QB + row) * DIMS + col];
        unsigned short h0 = f2bf(v.x), h1 = f2bf(v.y), h2 = f2bf(v.z), h3 = f2bf(v.w);
        unsigned short l0 = f2bf(v.x - __uint_as_float((unsigned)h0 << 16));
        unsigned short l1 = f2bf(v.y - __uint_as_float((unsigned)h1 << 16));
        unsigned short l2 = f2bf(v.z - __uint_as_float((unsigned)h2 << 16));
        unsigned short l3 = f2bf(v.w - __uint_as_float((unsigned)h3 << 16));
        uint2 hp; hp.x = (unsigned)h0 | ((unsigned)h1 << 16); hp.y = (unsigned)h2 | ((unsigned)h3 << 16);
        uint2 lp; lp.x = (unsigned)l0 | ((unsigned)l1 << 16); lp.y = (unsigned)l2 | ((unsigned)l3 << 16);
        *(uint2*)&qhi[row * LSTR + col] = hp;
        *(uint2*)&qlo[row * LSTR + col] = lp;
    }
    __syncthreads();

    const unsigned short* aHi = qhi + (qt * 16 + (lane & 15)) * LSTR;
    const unsigned short* aLo = qlo + (qt * 16 + (lane & 15)) * LSTR;
    const unsigned short* bHi = thi + (strm * 16 + (lane & 15)) * LSTR;
    const unsigned short* bLo = tlo + (strm * 16 + (lane & 15)) * LSTR;

    float bd[8][5]; int bi[8][5];
#pragma unroll
    for (int r = 0; r < 8; ++r)
#pragma unroll
        for (int j = 0; j < 5; ++j) { bd[r][j] = __FLT_MAX__; bi[r][j] = 0x7FFFFFFF; }

    const int nstreambase = nb * (NTRAIN / FB_NBLK) + strm * 2048;

    for (int it = 0; it < FB_TILES; ++it) {
        const int nbase = nstreambase + it * 16;
#pragma unroll 4
        for (int i = 0; i < 32; ++i) {
            int e = (tid & 63) * 4 + i * 256;
            int row = e >> 9, col = e & 511;
            float4 v = *(const float4*)&xt[(size_t)(nbase + row) * DIMS + col];
            unsigned short h0 = f2bf(v.x), h1 = f2bf(v.y), h2 = f2bf(v.z), h3 = f2bf(v.w);
            unsigned short l0 = f2bf(v.x - __uint_as_float((unsigned)h0 << 16));
            unsigned short l1 = f2bf(v.y - __uint_as_float((unsigned)h1 << 16));
            unsigned short l2 = f2bf(v.z - __uint_as_float((unsigned)h2 << 16));
            unsigned short l3 = f2bf(v.w - __uint_as_float((unsigned)h3 << 16));
            int base = (strm * 16 + row) * LSTR + col;
            uint2 hp; hp.x = (unsigned)h0 | ((unsigned)h1 << 16); hp.y = (unsigned)h2 | ((unsigned)h3 << 16);
            uint2 lp; lp.x = (unsigned)l0 | ((unsigned)l1 << 16); lp.y = (unsigned)l2 | ((unsigned)l3 << 16);
            *(uint2*)&thi[base] = hp;
            *(uint2*)&tlo[base] = lp;
        }
        if (it + 1 < FB_TILES)
            __builtin_prefetch(&xt[(size_t)(nbase + 16) * DIMS + (tid & 63) * 128], 0, 1);
        __syncthreads();

        v8f c = {0.f, 0.f, 0.f, 0.f, 0.f, 0.f, 0.f, 0.f};
#pragma unroll
        for (int k = 0; k < 16; ++k) {
            v16bf ah = ld_frag(aHi, k * 32 + half * 8,  k * 32 + 16 + half * 8);
            v16bf al = ld_frag(aLo, k * 32 + half * 8,  k * 32 + 16 + half * 8);
            v16bf bh = ld_frag(bHi, k * 32 + half * 8,  k * 32 + 16 + half * 8);
            v16bf bl = ld_frag(bLo, k * 32 + half * 8,  k * 32 + 16 + half * 8);
            c = __builtin_amdgcn_wmma_f32_16x16x32_bf16(false, ah, false, bh, (short)0, c, false, false);
            c = __builtin_amdgcn_wmma_f32_16x16x32_bf16(false, ah, false, bl, (short)0, c, false, false);
            c = __builtin_amdgcn_wmma_f32_16x16x32_bf16(false, al, false, bh, (short)0, c, false, false);
        }

        const int nIdx = nbase + (lane & 15);
        const float t2 = tn2[nIdx];
#pragma unroll
        for (int r = 0; r < 8; ++r) {
            float s = fmaf(-2.0f, c[r], t2);
            if (s < bd[r][4]) {
                bd[r][4] = s; bi[r][4] = nIdx;
#pragma unroll
                for (int j = 4; j >= 1; --j) {
                    if (bd[r][j] < bd[r][j - 1]) {
                        float td = bd[r][j]; bd[r][j] = bd[r][j - 1]; bd[r][j - 1] = td;
                        int   ti = bi[r][j]; bi[r][j] = bi[r][j - 1]; bi[r][j - 1] = ti;
                    }
                }
            }
        }
        __syncthreads();
    }

    float* sd = (float*)smem;
    int*   si = (int*)(smem + 25600);
#pragma unroll
    for (int r = 0; r < 8; ++r)
#pragma unroll
        for (int j = 0; j < 5; ++j) {
            int o = ((w * 32 + lane) * 8 + r) * 5 + j;
            sd[o] = bd[r][j]; si[o] = bi[r][j];
        }
    __syncthreads();

    if (lane < 16) {
        const int m = lane, hh = m >> 3, r = m & 7;
        float td[5]; int ti[5];
#pragma unroll
        for (int j = 0; j < 5; ++j) { td[j] = __FLT_MAX__; ti[j] = 0x7FFFFFFF; }
        for (int sl = 0; sl < 16; ++sl) {
            int L = hh * 16 + sl;
#pragma unroll
            for (int j = 0; j < 5; ++j) {
                int o = ((w * 32 + L) * 8 + r) * 5 + j;
                ins5(sd[o], si[o], td, ti);
            }
        }
        const int qg = bq * FB_QB + qt * 16 + m;
        const int split = nb * 2 + strm;
#pragma unroll
        for (int j = 0; j < 5; ++j) {
            candD[(qg * NSPLIT + split) * 5 + j] = td[j];
            candI[(qg * NSPLIT + split) * 5 + j] = ti[j];
        }
    }
}

// ---- kernel: final merge across splits + majority vote ----
__global__ __launch_bounds__(256) void knn_vote(const float* __restrict__ candD,
                                                const int* __restrict__ candI,
                                                const int* __restrict__ y,
                                                float* __restrict__ out) {
    int q = blockIdx.x * blockDim.x + threadIdx.x;
    if (q >= NQ) return;
    float td[5]; int ti[5];
#pragma unroll
    for (int j = 0; j < 5; ++j) { td[j] = __FLT_MAX__; ti[j] = 0x7FFFFFFF; }
    const float* dp = candD + (size_t)q * NSPLIT * 5;
    const int*   ip = candI + (size_t)q * NSPLIT * 5;
    for (int s = 0; s < NSPLIT * 5; ++s) ins5(dp[s], ip[s], td, ti);
    int lab[5];
#pragma unroll
    for (int j = 0; j < 5; ++j) lab[j] = y[ti[j]];
    int bestC = -1, bestLab = 0x7FFFFFFF;
#pragma unroll
    for (int j = 0; j < 5; ++j) {
        int c = 0;
#pragma unroll
        for (int k = 0; k < 5; ++k) c += (lab[k] == lab[j]) ? 1 : 0;
        if (c > bestC || (c == bestC && lab[j] < bestLab)) { bestC = c; bestLab = lab[j]; }
    }
    out[q] = (float)bestLab;
}

extern "C" void kernel_launch(void* const* d_in, const int* in_sizes, int n_in,
                              void* d_out, int out_size, void* d_ws, size_t ws_size,
                              hipStream_t stream) {
    const float* x  = (const float*)d_in[0];
    const float* xt = (const float*)d_in[1];
    const int*   y  = (const int*)d_in[2];
    float* out = (float*)d_out;

    char* ws = (char*)d_ws;
    float* tn2   = (float*)ws;                                   // 256 KB
    float* candD = (float*)(ws + 262144);                        // 1.31 MB
    int*   candI = (int*)(ws + 1572864);                         // 1.31 MB
    unsigned short* xthi = (unsigned short*)(ws + 2883584);      // 64 MB
    unsigned short* xtlo = (unsigned short*)(ws + 2883584 + 67108864);

    const size_t NEED_FAST = 2883584ull + 2ull * 67108864ull;    // ~137 MB

    train_norms<<<NTRAIN / 8, 256, 0, stream>>>(xt, tn2);
    if (ws_size >= NEED_FAST) {
        conv_train<<<(NTRAIN * DIMS) / 1024, 256, 0, stream>>>(xt, xthi, xtlo);
        knn_gemm_topk_fast<<<32 * FP_NBLK, 128, 0, stream>>>(x, xthi, xtlo, tn2, candD, candI);
    } else {
        knn_gemm_topk_fb<<<64 * FB_NBLK, 128, 0, stream>>>(x, xt, tn2, candD, candI);
    }
    knn_vote<<<(NQ + 255) / 256, 256, 0, stream>>>(candD, candI, y, out);
}
